// SelfAttention_773094113877
// MI455X (gfx1250) — matrix operand order repaired and checked by reading, not compiled, not run on hardware
//
#include <hip/hip_runtime.h>

#define EMBED 1024
#define HEADS 16
#define HDIM  64
#define NB    2
#define SEQ   2048

typedef __attribute__((ext_vector_type(16))) _Float16 v16h;
typedef __attribute__((ext_vector_type(8)))  _Float16 v8h;
typedef __attribute__((ext_vector_type(8)))  float    v8f;
typedef __attribute__((ext_vector_type(4)))  int      v4i;

typedef __attribute__((address_space(1))) v4i gv4i;   // global
typedef __attribute__((address_space(3))) v4i lv4i;   // LDS

#if defined(__has_builtin)
#if __has_builtin(__builtin_amdgcn_global_load_async_to_lds_b128) && \
    __has_builtin(__builtin_amdgcn_s_wait_asynccnt)
#define USE_ASYNC_LDS 1
#endif
#endif
#ifndef USE_ASYNC_LDS
#define USE_ASYNC_LDS 0
#endif

static __device__ __forceinline__ v8f wmma_f16(v16h a, v16h b, v8f c) {
  // D = A(16x32 f16) * B(32x16 f16) + C(16x16 f32)
  return __builtin_amdgcn_wmma_f32_16x16x32_f16(
      /*neg_a=*/false, a, /*neg_b=*/false, b,
      /*c_mod=*/(short)0, c, /*reuse_a=*/false, /*reuse_b=*/false);
}

static __device__ __forceinline__ v16h cat8(v8h lo, v8h hi) {
  return __builtin_shufflevector(lo, hi, 0,1,2,3,4,5,6,7,8,9,10,11,12,13,14,15);
}

// 16-byte global -> LDS copy (async on gfx1250 when the builtin is available)
static __device__ __forceinline__ void cp16_g2l(const _Float16* g, _Float16* l) {
#if USE_ASYNC_LDS
  __builtin_amdgcn_global_load_async_to_lds_b128((gv4i*)g, (lv4i*)l, 0, 0);
#else
  *(v8h*)l = *(const v8h*)g;
#endif
}

static __device__ __forceinline__ void wait_async_le2() {
#if USE_ASYNC_LDS
  __builtin_amdgcn_s_wait_asynccnt(2);
#endif
}
static __device__ __forceinline__ void wait_async_0() {
#if USE_ASYNC_LDS
  __builtin_amdgcn_s_wait_asynccnt(0);
#endif
}

// ---------------------------------------------------------------------------
// Phase 1: per-head projection  out[m][d] = sum_k x[m][k] * W[d][k]   (x @ W^T)
// transposed==0: out layout [n][h][l][d]   (Q, K)
// transposed==1: out layout [n][h][d][l]   (V)
// ---------------------------------------------------------------------------
__global__ __launch_bounds__(256)
void proj_f16_kernel(const float* __restrict__ X, const float* __restrict__ W,
                     _Float16* __restrict__ out, int transposed)
{
  __shared__ _Float16 Wl[HDIM * HDIM];
  const int tid = threadIdx.x;
  for (int i = tid; i < HDIM * HDIM; i += 256) Wl[i] = (_Float16)W[i];
  __syncthreads();

  const int lane = tid & 31;
  const int wave = tid >> 5;
  const int col  = lane & 15;
  const int sub  = lane >> 4;

  const int tile = blockIdx.x * 8 + wave;      // 0..4095 = n*H*(L/16)
  const int n  = tile >> 11;
  const int h  = (tile >> 7) & 15;
  const int l0 = (tile & 127) << 4;

  v16h bw[4][2];
  #pragma unroll
  for (int t = 0; t < 4; ++t) {
    #pragma unroll
    for (int c = 0; c < 2; ++c) {
      const int d = t * 16 + col;
      const int k = c * 32 + 8 * sub;
      const v8h* p = (const v8h*)&Wl[d * HDIM + k];
      bw[t][c] = cat8(p[0], p[2]);
    }
  }

  const float* xp = X + (size_t)(n * SEQ + (l0 + col)) * EMBED + h * HDIM;
  v16h ax[2];
  #pragma unroll
  for (int c = 0; c < 2; ++c) {
    const float4* lo = (const float4*)(xp + c * 32 + 8 * sub);
    const float4* hi = (const float4*)(xp + c * 32 + 16 + 8 * sub);
    float4 a0 = lo[0], a1 = lo[1], b0 = hi[0], b1 = hi[1];
    v16h a;
    a[0]=(_Float16)a0.x; a[1]=(_Float16)a0.y; a[2]=(_Float16)a0.z; a[3]=(_Float16)a0.w;
    a[4]=(_Float16)a1.x; a[5]=(_Float16)a1.y; a[6]=(_Float16)a1.z; a[7]=(_Float16)a1.w;
    a[8]=(_Float16)b0.x; a[9]=(_Float16)b0.y; a[10]=(_Float16)b0.z; a[11]=(_Float16)b0.w;
    a[12]=(_Float16)b1.x; a[13]=(_Float16)b1.y; a[14]=(_Float16)b1.z; a[15]=(_Float16)b1.w;
    ax[c] = a;
  }

  const size_t hb = (size_t)(n * HEADS + h);
  #pragma unroll
  for (int t = 0; t < 4; ++t) {
    v8f acc = {};
    acc = wmma_f16(ax[0], bw[t][0], acc);
    acc = wmma_f16(ax[1], bw[t][1], acc);
    #pragma unroll
    for (int r = 0; r < 8; ++r) {
      const int rr = l0 + 8 * sub + r;
      const int dd = t * 16 + col;
      const size_t off = transposed ? (hb * HDIM + dd) * SEQ + rr
                                    : (hb * SEQ + rr) * HDIM + dd;
      out[off] = (_Float16)acc[r];
    }
  }
}

// ---------------------------------------------------------------------------
// Phase 2: flash attention. Block = 8 waves sharing one (n,h); K/V tiles are
// staged cooperatively into double-buffered LDS via async global->LDS loads.
// One wave owns a 16-query tile. S^T = K*Q^T (its C-layout == P's A-layout).
// ---------------------------------------------------------------------------
__global__ __launch_bounds__(256)
void flash_attn_kernel(const _Float16* __restrict__ Qp,
                       const _Float16* __restrict__ Kp,
                       const _Float16* __restrict__ Vt,   // [n][h][d][key]
                       const int* __restrict__ mask,
                       float* __restrict__ out)
{
  __shared__ _Float16 Kl[2][32 * HDIM];   // [key_local][d]   4 KB x2
  __shared__ _Float16 Vl[2][HDIM * 32];   // [d][key_local]   4 KB x2

  const int tid  = threadIdx.x;
  const int lane = tid & 31;
  const int wave = tid >> 5;
  const int col  = lane & 15;
  const int sub  = lane >> 4;

  // block -> (n, h, 8 consecutive q-tiles)
  const int hbid = blockIdx.x >> 4;           // 0..31
  const int qgrp = blockIdx.x & 15;           // 0..15
  const int n  = hbid >> 4;
  const int h  = hbid & 15;
  const int q0 = (qgrp * 8 + wave) * 16;

  const size_t hoff = (size_t)(n * HEADS + h) * SEQ * HDIM;
  const _Float16* Qh = Qp + hoff;
  const _Float16* Kh = Kp + hoff;
  const _Float16* Vh = Vt + hoff;
  const int* mrow = mask + (size_t)n * SEQ * SEQ + (size_t)(q0 + col) * SEQ;

  // cooperative tile-load addressing (1 x b128 per thread per tile)
  const int krow = tid >> 3, kseg = tid & 7;  // K: 32 rows x 8 segs
  const int vrow = tid >> 2, vseg = tid & 3;  // V: 64 rows x 4 segs

  // Q^T as B-operand, loaded once
  v16h bq[2];
  #pragma unroll
  for (int c = 0; c < 2; ++c) {
    const v8h* p = (const v8h*)(Qh + (size_t)(q0 + col) * HDIM + c * 32 + 8 * sub);
    bq[c] = cat8(p[0], p[2]);
  }

  v8f o[4] = {v8f{}, v8f{}, v8f{}, v8f{}};
  float mstat = -1e30f, lstat = 0.0f;
  const float scale = 0.03125f;               // 1/sqrt(EMBED)

  // prologue: stage tile 0 into buffer 0
  cp16_g2l(Kh + (size_t)krow * HDIM + kseg * 8,      &Kl[0][krow * HDIM + kseg * 8]);
  cp16_g2l(Vh + (size_t)vrow * SEQ + vseg * 8,       &Vl[0][vrow * 32   + vseg * 8]);

  for (int kb = 0; kb < SEQ / 32; ++kb) {
    const int key0 = kb * 32;
    const int buf  = kb & 1;

    if (kb + 1 < SEQ / 32) {                  // stage next tile, then drain current
      const int nk = key0 + 32;
      cp16_g2l(Kh + (size_t)(nk + krow) * HDIM + kseg * 8, &Kl[buf ^ 1][krow * HDIM + kseg * 8]);
      cp16_g2l(Vh + (size_t)vrow * SEQ + nk + vseg * 8,    &Vl[buf ^ 1][vrow * 32   + vseg * 8]);
      wait_async_le2();
    } else {
      wait_async_0();
    }
    __syncthreads();

    __builtin_prefetch(mrow + key0 + 32, 0, 0);   // mask stream (dominant HBM traffic)

    // S^T tiles from LDS: A = K rows (16 keys x 64d), B = Q^T
    v8f st[2];
    #pragma unroll
    for (int jt = 0; jt < 2; ++jt) {
      const _Float16* kr = &Kl[buf][(jt * 16 + col) * HDIM];
      v8f acc = {};
      #pragma unroll
      for (int c = 0; c < 2; ++c) {
        const v8h* lo = (const v8h*)(kr + c * 32 + 8 * sub);
        const v8h* hi = (const v8h*)(kr + c * 32 + 16 + 8 * sub);
        acc = wmma_f16(cat8(lo[0], hi[0]), bq[c], acc);
      }
      st[jt] = acc;
    }

    // mask + scale: S^T row = key0 + jt*16 + 8*sub + r, col = query q0+col
    float p0[8], p1[8];
    {
      const int4* m0 = (const int4*)(mrow + key0 + 8 * sub);
      const int4* m1 = (const int4*)(mrow + key0 + 16 + 8 * sub);
      int4 a = m0[0], b = m0[1], c = m1[0], d = m1[1];
      const int mm0[8] = {a.x,a.y,a.z,a.w, b.x,b.y,b.z,b.w};
      const int mm1[8] = {c.x,c.y,c.z,c.w, d.x,d.y,d.z,d.w};
      #pragma unroll
      for (int r = 0; r < 8; ++r) {
        p0[r] = mm0[r] ? st[0][r] * scale : -1e20f;
        p1[r] = mm1[r] ? st[1][r] * scale : -1e20f;
      }
    }

    // online softmax; lane owns query q0+col (duplicated across halves)
    float tmax = -1e30f;
    #pragma unroll
    for (int r = 0; r < 8; ++r) tmax = fmaxf(tmax, fmaxf(p0[r], p1[r]));
    tmax = fmaxf(tmax, __shfl_xor(tmax, 16));
    const float mnew  = fmaxf(mstat, tmax);
    const float alpha = __expf(mstat - mnew);

    v16h pa;                                   // P already in A-operand layout
    float psum = 0.0f;
    #pragma unroll
    for (int r = 0; r < 8; ++r) {
      const float e0 = __expf(p0[r] - mnew);
      const float e1 = __expf(p1[r] - mnew);
      psum += e0 + e1;
      pa[r]     = (_Float16)e0;
      pa[r + 8] = (_Float16)e1;
    }
    psum += __shfl_xor(psum, 16);
    lstat = lstat * alpha + psum;
    mstat = mnew;

    #pragma unroll
    for (int r = 0; r < 8; ++r) {
      const float ar = __shfl(alpha, 8 * sub + r);
      #pragma unroll
      for (int t = 0; t < 4; ++t) o[t][r] *= ar;
    }

    // O += P * V from LDS
    #pragma unroll
    for (int t = 0; t < 4; ++t) {
      const v8h* vp = (const v8h*)(&Vl[buf][(t * 16 + col) * 32 + 8 * sub]);
      o[t] = wmma_f16(pa, cat8(vp[0], vp[2]), o[t]);
    }

    __syncthreads();   // all reads of this buffer done before it is re-filled
  }

  // normalize + store f32 output [n][q][h*64+d]
  #pragma unroll
  for (int r = 0; r < 8; ++r) {
    const float lr  = __shfl(lstat, 8 * sub + r);
    const float inv = 1.0f / lr;
    const int q = q0 + 8 * sub + r;
    float* orow = out + (size_t)(n * SEQ + q) * EMBED + h * HDIM;
    #pragma unroll
    for (int t = 0; t < 4; ++t) orow[t * 16 + col] = o[t][r] * inv;
  }
}

// ---------------------------------------------------------------------------
extern "C" void kernel_launch(void* const* d_in, const int* in_sizes, int n_in,
                              void* d_out, int out_size, void* d_ws, size_t ws_size,
                              hipStream_t stream) {
  (void)in_sizes; (void)n_in; (void)out_size; (void)ws_size;
  const float* values  = (const float*)d_in[0];
  const float* keys    = (const float*)d_in[1];
  const float* queries = (const float*)d_in[2];
  const int*   mask    = (const int*)d_in[3];
  const float* Wv      = (const float*)d_in[4];
  const float* Wk      = (const float*)d_in[5];
  const float* Wq      = (const float*)d_in[6];

  const size_t per = (size_t)NB * HEADS * SEQ * HDIM;  // 4M halfs = 8 MB each
  _Float16* Qp = (_Float16*)d_ws;
  _Float16* Kp = Qp + per;
  _Float16* Vt = Kp + per;

  dim3 blk(256);
  proj_f16_kernel<<<512, blk, 0, stream>>>(queries, Wq, Qp, 0);
  proj_f16_kernel<<<512, blk, 0, stream>>>(keys,    Wk, Kp, 0);
  proj_f16_kernel<<<512, blk, 0, stream>>>(values,  Wv, Vt, 1);
  flash_attn_kernel<<<512, blk, 0, stream>>>(Qp, Kp, Vt, mask, (float*)d_out);
}
